// DistanceBias_9878424781371
// MI455X (gfx1250) — matrix operand order repaired and checked by reading, not compiled, not run on hardware
//
#include <hip/hip_runtime.h>
#include <hip/hip_bf16.h>
#include <math.h>

typedef __attribute__((ext_vector_type(16))) _Float16 v16h;
typedef __attribute__((ext_vector_type(8)))  _Float16 v8h;
typedef __attribute__((ext_vector_type(8)))  float    v8f;
typedef __attribute__((ext_vector_type(4)))  float    v4f;

#define B_  2
#define N_  1024
#define P_  768
#define L_  256
#define K_  128
#define H_  32

#define NLL (B_*L_*(L_/16))        // 8192  : rows i<L,   cols 16j-blocks
#define NLP (B_*P_*(L_/16))        // 24576 : rows prot i, cols 16j-blocks
#define NPL (B_*L_*(P_/16))        // 24576 : rows lig j,  cols 16ip-blocks (transposed block)
#define NTILES (NLL+NLP+NPL)       // 57344

__device__ __forceinline__ float fast_rcp(float x) {
#if __has_builtin(__builtin_amdgcn_rcpf)
    return __builtin_amdgcn_rcpf(x);
#else
    return 1.0f / x;
#endif
}

__device__ __forceinline__ float fast_exp2(float x) {
#if __has_builtin(__builtin_amdgcn_exp2f)
    return __builtin_amdgcn_exp2f(x);      // raw v_exp_f32 (2^x)
#else
    return exp2f(x);
#endif
}

// Branch-free tanh-form GELU. f16 WMMA already limits accuracy to ~1e-3 rel
// over a 128-deep dot, so the ~1e-3 tanh-GELU approximation is within noise.
__device__ __forceinline__ float gelu_fast(float x) {
    float t = x * x;
    float a = __builtin_fmaf(0.044715f, t, 1.0f);     // 1 + 0.044715 x^2
#if __has_builtin(__builtin_amdgcn_tanhf)
    float u  = (x * 0.7978845608028654f) * a;         // sqrt(2/pi) x (1+..)
    float th = __builtin_amdgcn_tanhf(u);             // single v_tanh_f32
    return x * __builtin_fmaf(th, 0.5f, 0.5f);
#else
    float b = x * -2.3022081f;                        // -2*sqrt(2/pi)*log2(e)
    float e = fast_exp2(a * b);                       // 2^(-2u*log2e) = e^(-2u)
    return x * fast_rcp(e + 1.0f);                    // x * sigmoid(2u)
#endif
}

// A-fragment (16-bit, 16x32) K index for packed element e = 2v+p, lane group g.
// Contiguous order (matches B staging): lanes 0-15 K=0..15, lanes 16-31 K=16..31.
__device__ __forceinline__ int a_kidx(int e, int g) {
    int v = e >> 1, p = e & 1;
    return 16 * g + 2 * v + p;
}

#define WMMA_F16(A, Bf, C) \
    __builtin_amdgcn_wmma_f32_16x16x32_f16(false, (A), false, (Bf), (short)0, (C), false, false)

extern "C" __global__ void db_zero_pp(float* __restrict__ out) {
    size_t idx = (size_t)blockIdx.x * 256 + threadIdx.x;   // one float4 each
    int   c4    = (int)(idx % (P_/4));
    size_t r    = idx / (P_/4);
    int   row   = (int)(r % P_);
    int   plane = (int)(r / P_);                           // b*H + h, 0..63
    float* p = out + ((size_t)plane * N_ + L_ + row) * N_ + L_ + c4 * 4;
    v4f z = {};
    *(v4f*)p = z;
}

extern "C" __global__ __launch_bounds__(256)
void db_main(const float* __restrict__ pos,
             const int*   __restrict__ etypes,
             const float* __restrict__ means,
             const float* __restrict__ stds,
             const float* __restrict__ mul_w,
             const float* __restrict__ bias_w,
             const float* __restrict__ ow1,
             const float* __restrict__ ob1,
             const float* __restrict__ ow2,
             const float* __restrict__ ob2,
             const float* __restrict__ vw1,
             const float* __restrict__ vb1,
             const float* __restrict__ vw2,
             const float* __restrict__ vb2,
             float* __restrict__ out)
{
    // Weights pre-swizzled into WMMA B-fragment order:
    // fragment f, lane l, 16 contiguous halves per lane (two ds_load_b128).
    __shared__ __attribute__((aligned(32))) _Float16 sW1 [32*32*16];  // 32KB: 4 kb x 8 nb
    __shared__ __attribute__((aligned(32))) _Float16 sW2 [ 8*32*16];  // 8KB : 4 kb x 2 nb
    __shared__ __attribute__((aligned(32))) _Float16 sVW2[ 8*32*16];  // 8KB
    __shared__ __attribute__((aligned(32))) _Float16 sH[8][K_*16];    // per-wave staging, feature-major
    __shared__ __attribute__((aligned(16))) v4f sGP[K_];   // {mean, inv_std*c, coef, -}
    __shared__ __attribute__((aligned(16))) v4f sVP[K_];   // {vw1.x, vw1.y, vw1.z, vb1}
    __shared__ float sOB1[K_];
    __shared__ float sOB2[H_], sVB2[H_];

    const int tid = threadIdx.x;

    // ---- stage weights (f32 -> f16, B-fragment layout) ----
    for (int idx = tid; idx < 32*512; idx += 256) {
        int e = idx & 15, l = (idx >> 4) & 31, f = idx >> 9;
        int kb = f >> 3, nb = f & 7;
        int k = 32*kb + 16*(l >> 4) + e;      // B layout: lanes 0-15 K=0..15, 16-31 K=16..31
        int n = 16*nb + (l & 15);
        sW1[idx] = (_Float16)ow1[k*K_ + n];
    }
    for (int idx = tid; idx < 8*512; idx += 256) {
        int e = idx & 15, l = (idx >> 4) & 31, f = idx >> 9;
        int kb = f >> 1, nb = f & 1;
        int k = 32*kb + 16*(l >> 4) + e;
        int n = 16*nb + (l & 15);
        sW2[idx]  = (_Float16)ow2[k*H_ + n];
        sVW2[idx] = (_Float16)vw2[k*H_ + n];
    }
    for (int idx = tid; idx < K_; idx += 256) {
        float s  = fabsf(stds[idx]) + 1e-5f;
        float is = 1.0f / s;
        // exp(-0.5 z^2) = 2^(-(z*sqrt(0.5*log2e))^2): fold into inv_std.
        v4f gp = { means[idx], is * 0.84932181f, 0.3989427f * is, 0.0f };
        sGP[idx] = gp;
        v4f vp = { vw1[idx], vw1[K_ + idx], vw1[2*K_ + idx], vb1[idx] };
        sVP[idx] = vp;
        sOB1[idx] = ob1[idx];
    }
    if (tid < H_) { sOB2[tid] = ob2[tid]; sVB2[tid] = vb2[tid]; }
    __syncthreads();

    const int wave = tid >> 5;
    const int lane = tid & 31;
    const int m = lane & 15;      // A-row / C-column index
    const int g = lane >> 4;      // lane group
    const int tile = blockIdx.x * 8 + wave;
    if (tile >= NTILES) return;

    // ---- tile decomposition ----
    int kind, b, ri, cb;
    if (tile < NLL) {
        int t = tile; kind = 0;
        b = t / (L_*16); int r = t % (L_*16); ri = r / 16; cb = r % 16;
    } else if (tile < NLL + NLP) {
        int t = tile - NLL; kind = 1;
        b = t / (P_*16); int r = t % (P_*16); ri = r / 16; cb = r % 16;
    } else {
        int t = tile - NLL - NLP; kind = 2;
        b = t / (L_*(P_/16)); int r = t % (L_*(P_/16)); ri = r / (P_/16); cb = r % (P_/16);
    }

    // ---- per-lane edge geometry (edge index within tile = m) ----
    int rowPt, colPt, etIdx; float dsign;
    if (kind == 0) {                 // ef[b, i, j]  (i<L, j=lig)
        int i = ri, j = cb*16 + m;
        rowPt = b*N_ + i;        colPt = b*N_ + j;
        etIdx = (b*N_ + i)*N_ + j;                 dsign =  1.0f;
    } else if (kind == 1) {          // ef[b, L+i, j] (prot row)
        int i = ri, j = cb*16 + m;
        rowPt = b*N_ + L_ + i;   colPt = b*N_ + j;
        etIdx = (b*N_ + L_ + i)*N_ + j;            dsign =  1.0f;
    } else {                         // ef[b, jl, L+ip] = ef_lp[b,ip,jl] + nl(-d)
        int jl = ri, ip = cb*16 + m;
        rowPt = b*N_ + L_ + ip;  colPt = b*N_ + jl;
        etIdx = (b*N_ + L_ + ip)*N_ + jl;          dsign = -1.0f;
    }
    float rx = pos[rowPt*3+0], ry = pos[rowPt*3+1], rz = pos[rowPt*3+2];
    float cx = pos[colPt*3+0], cy = pos[colPt*3+1], cz = pos[colPt*3+2];
    float ddx = cx-rx, ddy = cy-ry, ddz = cz-rz;   // dlm = lig[col] - rowpos
    float dist = fast_rcp(__builtin_fmaf(ddx, ddx,
                          __builtin_fmaf(ddy, ddy,
                          __builtin_fmaf(ddz, ddz, 1.0f))));
    int   et = etypes[etIdx];
    float t0 = __builtin_fmaf(dist, mul_w[et], bias_w[et]);

    // ---- Gaussian RBF directly in A-fragment layout ----
    v16h aG[4];
    #pragma unroll
    for (int kb = 0; kb < 4; ++kb) {
        #pragma unroll
        for (int e = 0; e < 16; ++e) {
            int k = 32*kb + a_kidx(e, g);
            v4f gp = sGP[k];                       // broadcast ds_load_b128
            float x = (t0 - gp.x) * gp.y;          // pre-scaled by sqrt(0.5*log2e)
            aG[kb][e] = (_Float16)(gp.z * fast_exp2(-(x * x)));
        }
    }

    // ---- GEMM1: [16x128] @ ow1[128x128], gelu, stage feature-major to LDS ----
    _Float16* myH = sH[wave];
    #pragma unroll
    for (int nb = 0; nb < 8; ++nb) {
        v8f c = {};
        #pragma unroll
        for (int kb = 0; kb < 4; ++kb) {
            v16h bf = *(const v16h*)&sW1[((kb*8 + nb)*32 + lane)*16];
            c = WMMA_F16(aG[kb], bf, c);
        }
        float b1 = sOB1[16*nb + m];     // D: lane -> column N = m (= feature 16nb+m)
        v8h hp;
        #pragma unroll
        for (int v = 0; v < 8; ++v) {   // D: VGPR v -> row M = v + 8g (= edge)
            hp[v] = (_Float16)gelu_fast(c[v] + b1);
        }
        *(v8h*)&myH[(16*nb + m)*16 + 8*g] = hp;   // one ds_store_b128
    }

    // ---- A fragments for GEMM2 (from staging) and the geometric MLP ----
    v16h aH[4], aV[4];
    float ex = dsign*ddx, ey = dsign*ddy, ez = dsign*ddz;
    #pragma unroll
    for (int kb = 0; kb < 4; ++kb) {
        #pragma unroll
        for (int e = 0; e < 16; ++e) {
            int k = 32*kb + a_kidx(e, g);
            aH[kb][e] = myH[k*16 + m];
            v4f vp = sVP[k];                       // broadcast ds_load_b128
            float x = __builtin_fmaf(ex, vp.x,
                      __builtin_fmaf(ey, vp.y,
                      __builtin_fmaf(ez, vp.z, vp.w)));
            aV[kb][e] = (_Float16)gelu_fast(x);
        }
    }

    // ---- GEMM2 (H=32): both MLPs chained into ONE accumulator + fused store ----
    size_t rowOut, colOut;
    if (kind == 0)      { rowOut = (size_t)ri;       colOut = (size_t)cb*16; }
    else if (kind == 1) { rowOut = (size_t)(L_+ri);  colOut = (size_t)cb*16; }
    else                { rowOut = (size_t)ri;       colOut = (size_t)(L_ + cb*16); }

    #pragma unroll
    for (int nb = 0; nb < 2; ++nb) {
        v8f c = {};
        #pragma unroll
        for (int kb = 0; kb < 4; ++kb) {
            v16h bV = *(const v16h*)&sVW2[((kb*2 + nb)*32 + lane)*16];
            c = WMMA_F16(aV[kb], bV, c);
        }
        #pragma unroll
        for (int kb = 0; kb < 4; ++kb) {
            v16h bE = *(const v16h*)&sW2[((kb*2 + nb)*32 + lane)*16];
            c = WMMA_F16(aH[kb], bE, c);
        }
        int h = 16*nb + m;
        float badd = sOB2[h] + sVB2[h];
        float* op = out + (((size_t)(b*H_ + h)*N_ + rowOut)*N_ + colOut + 8*g);
        v4f w0 = { c[0]+badd, c[1]+badd, c[2]+badd, c[3]+badd };
        v4f w1 = { c[4]+badd, c[5]+badd, c[6]+badd, c[7]+badd };
        *(v4f*)(op)     = w0;           // rows M = v+8g are contiguous in j
        *(v4f*)(op + 4) = w1;
    }
}

extern "C" void kernel_launch(void* const* d_in, const int* in_sizes, int n_in,
                              void* d_out, int out_size, void* d_ws, size_t ws_size,
                              hipStream_t stream) {
    const float* pos    = (const float*)d_in[0];
    const int*   et     = (const int*)  d_in[1];
    // d_in[2] = protein_length (constant 768, compiled in)
    const float* means  = (const float*)d_in[3];
    const float* stds   = (const float*)d_in[4];
    const float* mul_w  = (const float*)d_in[5];
    const float* bias_w = (const float*)d_in[6];
    const float* ow1    = (const float*)d_in[7];
    const float* ob1    = (const float*)d_in[8];
    const float* ow2    = (const float*)d_in[9];
    const float* ob2    = (const float*)d_in[10];
    const float* vw1    = (const float*)d_in[11];
    const float* vb1    = (const float*)d_in[12];
    const float* vw2    = (const float*)d_in[13];
    const float* vb2    = (const float*)d_in[14];
    float* out = (float*)d_out;

    // protein-protein block of the output stays zero
    const size_t ppVec4 = (size_t)B_ * H_ * P_ * P_ / 4;   // 9,437,184
    db_zero_pp<<<dim3((unsigned)(ppVec4 / 256)), dim3(256), 0, stream>>>(out);

    db_main<<<dim3(NTILES / 8), dim3(256), 0, stream>>>(
        pos, et, means, stds, mul_w, bias_w,
        ow1, ob1, ow2, ob2, vw1, vb1, vw2, vb2, out);
}
